// AFNO2D_31095563223138
// MI455X (gfx1250) — compile-verified, matrix operand order checked by reading, not run on hardware
//
#include <hip/hip_runtime.h>

// ---------------------------------------------------------------------------
// AFNO2D for MI455X (gfx1250): rfft2 -> blockdiag complex MLP (WMMA f16) ->
// irfft2 + residual.  B=8,H=128,W=128,C=768, 8 blocks of 96, Wf=65.
// ---------------------------------------------------------------------------

#define BB   8
#define HH   128
#define WW   128
#define CC   768
#define NB   8
#define BS   96
#define WF   65          // W/2+1
#define PTS  (BB*HH*WF)  // 66560 frequency points
#define STR  132         // padded LDS row stride (float2) for FFT tiles

#define FRAG_HALFS_PER_BLOCK 36864   // 4 mats * 18 frags * 512 halfs
#define FRAG_BYTES_PER_BLOCK (FRAG_HALFS_PER_BLOCK * 2)

typedef __attribute__((ext_vector_type(16))) _Float16 v16h;
typedef __attribute__((ext_vector_type(8)))  float    v8f;
typedef int v4i_ __attribute__((vector_size(16)));   // b128 payload type

__device__ __forceinline__ unsigned rev7(unsigned v) { return __brev(v) >> 25; }

__device__ __forceinline__ float softshrink01(float v) {
    const float L = 0.01f;
    return v > L ? v - L : (v < -L ? v + L : 0.0f);
}

// 16 independent 128-point complex FFTs living in LDS (t[c*STR + idx]),
// input already bit-reversed.  128 threads -> 8 butterflies each per stage.
template <int SIGN>
__device__ void fft128_16(float2* t, int tid) {
    for (int s = 0; s < 7; ++s) {
        const int half = 1 << s;
        __syncthreads();
#pragma unroll 1
        for (int k = 0; k < 8; ++k) {
            int bf = tid + (k << 7);          // 0..1023
            int c  = bf >> 6;                 // transform 0..15
            int q  = bf & 63;                 // butterfly within transform
            int j  = q & (half - 1);
            int g  = q >> s;
            int i0 = (g << (s + 1)) + j;
            int i1 = i0 + half;
            float ang = (float)SIGN * 3.14159265358979323846f * (float)j / (float)half;
            float wi, wr;
            __sincosf(ang, &wi, &wr);
            float2 u = t[c * STR + i0];
            float2 v = t[c * STR + i1];
            float tr = v.x * wr - v.y * wi;
            float ti = v.x * wi + v.y * wr;
            t[c * STR + i0] = make_float2(u.x + tr, u.y + ti);
            t[c * STR + i1] = make_float2(u.x - tr, u.y - ti);
        }
    }
    __syncthreads();
}

// ---- pass 0: bake f32 weights into f16 WMMA-B fragment order (once) -------
// Blob layout per block n: fid = ((mat*3+kc)*6+jt), 512 halfs per frag
// (32 lanes x 16 halfs).  B 32x16 f16 layout: lane L holds N=L%16; half i
// maps to K = (i&7) + 16*(i>>3) + 8*(L>=16) within the 32-wide K chunk.
__global__ void __launch_bounds__(256)
prep_weights(const float* __restrict__ w1, const float* __restrict__ w2,
             _Float16* __restrict__ wblob) {
    int e = blockIdx.x * 256 + threadIdx.x;          // 0 .. NB*36864-1
    if (e >= NB * FRAG_HALFS_PER_BLOCK) return;
    int n   = e / FRAG_HALFS_PER_BLOCK;
    int r   = e % FRAG_HALFS_PER_BLOCK;
    int i   = r & 15;
    int ln  = (r >> 4) & 31;
    int fid = r >> 9;
    int jt  = fid % 6;
    int kc  = (fid / 6) % 3;
    int mat = fid / 18;                              // 0=w1r 1=w1i 2=w2r 3=w2i
    int kk  = (i & 7) + ((i >> 3) << 4) + ((ln >> 4) << 3);
    int K   = kc * 32 + kk;
    int N   = jt * 16 + (ln & 15);
    const float* ws = (mat < 2) ? w1 : w2;
    int ri = mat & 1;
    wblob[e] = (_Float16)ws[((size_t)(ri * NB + n) * BS + K) * BS + N];
}

// ---- pass 1: real FFT along W, keep bins 0..64, scale 1/sqrt(128) ----------
__global__ void __launch_bounds__(128)
fft_rows_fwd(const float* __restrict__ x, float2* __restrict__ spec) {
    __shared__ float2 t[16 * STR];
    const int bh  = blockIdx.x;          // b*128+h
    const int c0  = blockIdx.y * 16;
    const int tid = threadIdx.x;
    for (int e = tid; e < 128 * 16; e += 128) {
        int w = e >> 4, c = e & 15;
        float v = x[((size_t)bh * WW + w) * CC + c0 + c];
        t[c * STR + rev7(w)] = make_float2(v, 0.0f);
    }
    fft128_16<-1>(t, tid);
    const float sc = 0.08838834764831845f;  // 1/sqrt(128)
    for (int e = tid; e < WF * 16; e += 128) {
        int wf = e >> 4, c = e & 15;
        float2 v = t[c * STR + wf];
        spec[((size_t)bh * WF + wf) * CC + c0 + c] = make_float2(v.x * sc, v.y * sc);
    }
}

// ---- pass 2/4: complex FFT along H, in place, scale 1/sqrt(128) ------------
template <int SIGN>
__global__ void __launch_bounds__(128)
fft_cols(float2* __restrict__ spec) {
    __shared__ float2 t[16 * STR];
    const int bwf = blockIdx.x;          // b*65+wf
    const int b = bwf / WF, wf = bwf % WF;
    const int c0  = blockIdx.y * 16;
    const int tid = threadIdx.x;
    for (int e = tid; e < 128 * 16; e += 128) {
        int h = e >> 4, c = e & 15;
        float2 v = spec[(((size_t)b * HH + h) * WF + wf) * CC + c0 + c];
        t[c * STR + rev7(h)] = v;
    }
    fft128_16<SIGN>(t, tid);
    const float sc = 0.08838834764831845f;
    for (int e = tid; e < 128 * 16; e += 128) {
        int h = e >> 4, c = e & 15;
        float2 v = t[c * STR + h];
        spec[(((size_t)b * HH + h) * WF + wf) * CC + c0 + c] =
            make_float2(v.x * sc, v.y * sc);
    }
}

// ---- pass 3: block-diagonal complex 2-layer MLP via v_wmma_f32_16x16x32_f16
// One workgroup = 4 waves; each wave owns a 16-point M tile; all waves share
// the block-n weights, bulk-copied (async global->LDS when available) from the
// pre-swizzled f16 fragment blob.
__global__ void __launch_bounds__(128)
afno_mlp(float2* __restrict__ spec, const _Float16* __restrict__ wblob,
         const float* __restrict__ b1, const float* __restrict__ b2) {
    extern __shared__ v16h smemv[];
    _Float16* wfrag = (_Float16*)smemv;                 // 36864 halfs (4 mats)
    float*    bias  = (float*)((char*)smemv + 73728);   // 4*96 floats
    _Float16* o1st  = (_Float16*)((char*)smemv + 75264);// 4 waves * 2*16*96 halfs

    const int n    = blockIdx.y;
    const int tid  = threadIdx.x;
    const int lane = tid & 31;
    const int wave = tid >> 5;

    // --- bulk copy this block's fragment blob into LDS (72 KB) ---
    const _Float16* wsrc = wblob + (size_t)n * FRAG_HALFS_PER_BLOCK;
#if __has_builtin(__builtin_amdgcn_global_load_async_to_lds_b128)
    // CDNA5 async copy: no VGPR round trip, tracked with ASYNCcnt.
#pragma unroll 1
    for (int k = 0; k < 36; ++k) {
        int idx = tid + (k << 7);                       // 0..4607 (16B each)
        __builtin_amdgcn_global_load_async_to_lds_b128(
            (__attribute__((address_space(1))) v4i_*)(wsrc + idx * 8),
            (__attribute__((address_space(3))) v4i_*)(wfrag + idx * 8),
            0, 0);
    }
    asm volatile("s_wait_asynccnt 0" ::: "memory");
#else
    {
        const float4* s4 = (const float4*)wsrc;
        float4* d4 = (float4*)wfrag;
        for (int idx = tid; idx < 4608; idx += 128) d4[idx] = s4[idx];
    }
#endif
    for (int e = tid; e < 384; e += 128) {
        int mat = e / 96, o = e % 96;
        const float* bsrc = (mat < 2) ? b1 : b2;
        int ri = mat & 1;
        bias[e] = bsrc[(ri * NB + n) * BS + o];
    }
    __syncthreads();

    const v16h* fragv = (const v16h*)wfrag;   // frag fid -> fragv[fid*32 + lane]
    const int p0 = (blockIdx.x * 4 + wave) * 16;
    const int Nl = lane & 15;
    const int hi = (lane >> 4) << 3;          // +8 for upper half-wave

    // --- A fragments: xr, xi, -xi  (A 16x32 f16: lane holds M=lane%16) ---
    v16h ar[3], ai_[3], an[3];
    const size_t prow = (size_t)(p0 + Nl) * CC + n * BS;
#pragma unroll
    for (int kc = 0; kc < 3; ++kc) {
#pragma unroll
        for (int i = 0; i < 16; ++i) {
            int kk = (i & 7) + ((i >> 3) << 4) + hi;
            float2 v = spec[prow + kc * 32 + kk];
            ar[kc][i]  = (_Float16)v.x;
            ai_[kc][i] = (_Float16)v.y;
            an[kc][i]  = (_Float16)(-v.y);
        }
    }

    _Float16* my_r = o1st + wave * 3072;
    _Float16* my_i = my_r + 1536;

    // --- layer 1: o1r = relu(Ar*W1r + (-Ai)*W1i + b1r), o1i analogous ---
    // (f16 WMMA NEG only allows C-negate, so complex subtraction negates A.)
    for (int jt = 0; jt < 6; ++jt) {
        v8f cr, ci;
        float br = bias[0 * 96 + jt * 16 + Nl];
        float bi = bias[1 * 96 + jt * 16 + Nl];
#pragma unroll
        for (int r = 0; r < 8; ++r) { cr[r] = br; ci[r] = bi; }
#pragma unroll
        for (int kc = 0; kc < 3; ++kc) {
            v16h fw1r = fragv[((0 * 3 + kc) * 6 + jt) * 32 + lane];
            v16h fw1i = fragv[((1 * 3 + kc) * 6 + jt) * 32 + lane];
            cr = __builtin_amdgcn_wmma_f32_16x16x32_f16(false, ar[kc],  false, fw1r, (short)0, cr, false, false);
            cr = __builtin_amdgcn_wmma_f32_16x16x32_f16(false, an[kc],  false, fw1i, (short)0, cr, false, false);
            ci = __builtin_amdgcn_wmma_f32_16x16x32_f16(false, ai_[kc], false, fw1r, (short)0, ci, false, false);
            ci = __builtin_amdgcn_wmma_f32_16x16x32_f16(false, ar[kc],  false, fw1i, (short)0, ci, false, false);
        }
        // relu + stage to LDS in [M][N] order (D layout: M = reg + hi, N = lane%16)
#pragma unroll
        for (int r = 0; r < 8; ++r) {
            int M = hi + r;
            my_r[M * BS + jt * 16 + Nl] = (_Float16)fmaxf(cr[r], 0.0f);
            my_i[M * BS + jt * 16 + Nl] = (_Float16)fmaxf(ci[r], 0.0f);
        }
    }
    asm volatile("s_wait_dscnt 0" ::: "memory");   // wave-local LDS RAW fence

    // --- transpose-gather layer-1 outputs into A fragments ---
    v16h orr[3], oii[3], onn[3];
#pragma unroll
    for (int kc = 0; kc < 3; ++kc) {
#pragma unroll
        for (int i = 0; i < 16; ++i) {
            int kk = (i & 7) + ((i >> 3) << 4) + hi;
            int K  = kc * 32 + kk;
            _Float16 rv = my_r[Nl * BS + K];
            _Float16 iv = my_i[Nl * BS + K];
            orr[kc][i] = rv;
            oii[kc][i] = iv;
            onn[kc][i] = (_Float16)(-(float)iv);
        }
    }

    // --- layer 2 + softshrink, write back in place ---
    for (int jt = 0; jt < 6; ++jt) {
        v8f cr, ci;
        float br = bias[2 * 96 + jt * 16 + Nl];
        float bi = bias[3 * 96 + jt * 16 + Nl];
#pragma unroll
        for (int r = 0; r < 8; ++r) { cr[r] = br; ci[r] = bi; }
#pragma unroll
        for (int kc = 0; kc < 3; ++kc) {
            v16h fw2r = fragv[((2 * 3 + kc) * 6 + jt) * 32 + lane];
            v16h fw2i = fragv[((3 * 3 + kc) * 6 + jt) * 32 + lane];
            cr = __builtin_amdgcn_wmma_f32_16x16x32_f16(false, orr[kc], false, fw2r, (short)0, cr, false, false);
            cr = __builtin_amdgcn_wmma_f32_16x16x32_f16(false, onn[kc], false, fw2i, (short)0, cr, false, false);
            ci = __builtin_amdgcn_wmma_f32_16x16x32_f16(false, oii[kc], false, fw2r, (short)0, ci, false, false);
            ci = __builtin_amdgcn_wmma_f32_16x16x32_f16(false, orr[kc], false, fw2i, (short)0, ci, false, false);
        }
#pragma unroll
        for (int r = 0; r < 8; ++r) {
            int p = p0 + hi + r;
            spec[(size_t)p * CC + n * BS + jt * 16 + Nl] =
                make_float2(softshrink01(cr[r]), softshrink01(ci[r]));
        }
    }
}

// ---- pass 5: Hermitian-extend, inverse FFT along W, add residual -----------
__global__ void __launch_bounds__(128)
ifft_rows_out(const float2* __restrict__ spec, const float* __restrict__ x,
              float* __restrict__ y) {
    __shared__ float2 t[16 * STR];
    const int bh  = blockIdx.x;
    const int c0  = blockIdx.y * 16;
    const int tid = threadIdx.x;
    for (int e = tid; e < 128 * 16; e += 128) {
        int w = e >> 4, c = e & 15;
        float2 v;
        if (w <= 64) {
            v = spec[((size_t)bh * WF + w) * CC + c0 + c];
        } else {
            float2 u = spec[((size_t)bh * WF + (128 - w)) * CC + c0 + c];
            v = make_float2(u.x, -u.y);
        }
        t[c * STR + rev7(w)] = v;
    }
    fft128_16<1>(t, tid);
    const float sc = 0.08838834764831845f;
    for (int e = tid; e < 128 * 16; e += 128) {
        int w = e >> 4, c = e & 15;
        size_t idx = ((size_t)bh * WW + w) * CC + c0 + c;
        y[idx] = t[c * STR + w].x * sc + x[idx];
    }
}

// ---------------------------------------------------------------------------
extern "C" void kernel_launch(void* const* d_in, const int* in_sizes, int n_in,
                              void* d_out, int out_size, void* d_ws, size_t ws_size,
                              hipStream_t stream) {
    const float* x  = (const float*)d_in[0];
    const float* w1 = (const float*)d_in[1];
    const float* b1 = (const float*)d_in[2];
    const float* w2 = (const float*)d_in[3];
    const float* b2 = (const float*)d_in[4];
    float* y = (float*)d_out;

    const size_t specBytes = (size_t)PTS * CC * sizeof(float2);   // 409 MB
    const size_t blobBytes = (size_t)NB * FRAG_BYTES_PER_BLOCK;   // 576 KB
    if (ws_size < specBytes + blobBytes) return;

    float2*   spec  = (float2*)d_ws;
    _Float16* wblob = (_Float16*)((char*)d_ws + specBytes);

    dim3 blk(128);
    prep_weights<<<dim3((NB * FRAG_HALFS_PER_BLOCK + 255) / 256), dim3(256), 0, stream>>>(w1, w2, wblob);
    fft_rows_fwd<<<dim3(BB * HH, CC / 16), blk, 0, stream>>>(x, spec);
    fft_cols<-1><<<dim3(BB * WF, CC / 16), blk, 0, stream>>>(spec);

    const size_t smem = 73728 + 1536 + 4 * 3072 * sizeof(_Float16); // 99840 B
    afno_mlp<<<dim3(PTS / 64, NB), blk, smem, stream>>>(spec, wblob, b1, b2);

    fft_cols<1><<<dim3(BB * WF, CC / 16), blk, 0, stream>>>(spec);
    ifft_rows_out<<<dim3(BB * HH, CC / 16), blk, 0, stream>>>(spec, x, y);
}
